// MoE_17772574671183
// MI455X (gfx1250) — matrix operand order repaired and checked by reading, not compiled, not run on hardware
//
#include <hip/hip_runtime.h>
#include <math.h>

// ---------------- problem constants ----------------
#define NE   8          // experts
#define DIN  128        // input dim
#define HID  256        // expert hidden
#define NTOK (16*4096)  // T = B*N
#define PAD  8
#define SW_D (DIN+PAD)  // 136 halves  (row stride for [.,D] LDS tiles)
#define SW_H (HID+PAD)  // 264 halves  (row stride for [.,H] LDS tiles)

typedef __attribute__((ext_vector_type(16))) _Float16 v16h;
typedef __attribute__((ext_vector_type(8)))  float    v8f;

union FragH { v16h v; uint4 q[2]; };

// D = A(16x32 f16) * B(32x16 f16) + C(16x16 f32)
__device__ __forceinline__ v8f wmma16(v16h a, v16h b, v8f c) {
    return __builtin_amdgcn_wmma_f32_16x16x32_f16(false, a, false, b,
                                                  (short)0, c, false, false);
}

// A fragment: row = lane&15, K halves {kb+off..+7, kb+16+off..+7}, off = 8*(lane>=16)
__device__ __forceinline__ v16h loadA(const _Float16* row, int kb, int lane) {
    FragH f;
    const _Float16* p = row + kb + ((lane >> 4) << 3);
    f.q[0] = *reinterpret_cast<const uint4*>(p);
    f.q[1] = *reinterpret_cast<const uint4*>(p + 16);
    return f.v;
}
// B fragment from LDS weight stored transposed [N][K] (stride in halves)
__device__ __forceinline__ v16h loadB(const _Float16* mat, int stride, int nb,
                                      int kb, int lane) {
    FragH f;
    const _Float16* p = mat + (nb + (lane & 15)) * stride + kb + ((lane >> 4) << 3);
    f.q[0] = *reinterpret_cast<const uint4*>(p);
    f.q[1] = *reinterpret_cast<const uint4*>(p + 16);
    return f.v;
}

// ---- CDNA5 async global->LDS copy (ASYNCcnt-tracked, bypasses VGPRs) ----
// LDS operand: low 32 bits of the generic pointer == wave-relative LDS offset.
__device__ __forceinline__ void async_ld_b128(const _Float16* lds,
                                              const _Float16* base,
                                              uint32_t byteOff) {
    const uint32_t l = (uint32_t)(uintptr_t)lds;
    asm volatile("global_load_async_to_lds_b128 %0, %1, %2"
                 :: "v"(l), "v"(byteOff), "s"(base) : "memory");
}
__device__ __forceinline__ void wait_async0() {
    asm volatile("s_wait_asynccnt 0x0" ::: "memory");
}

// ---------------- prep: f32 -> f16 transpose-convert ----------------
// dst[b][n][k] = (f16) src[b][k*N + n]   (K x N row-major in, N x K row-major out)
__global__ __launch_bounds__(256)
void transcvt_kernel(const float* __restrict__ src, _Float16* __restrict__ dst,
                     int K, int N) {
    const int b = blockIdx.y;
    src += (size_t)b * K * N;
    dst += (size_t)b * K * N;
    const int idx = blockIdx.x * 256 + threadIdx.x;    // output-linear n*K + k
    if (idx < K * N) {
        const int n = idx / K, k = idx % K;
        dst[idx] = (_Float16)src[(size_t)k * N + n];
    }
}

// x (f32, coalesced) -> xh (f16)
__global__ __launch_bounds__(256)
void xcvt_kernel(const float* __restrict__ x, _Float16* __restrict__ xh) {
    const size_t i = ((size_t)blockIdx.x * 256 + threadIdx.x) * 8;
    const float4 a = *reinterpret_cast<const float4*>(x + i);
    const float4 b = *reinterpret_cast<const float4*>(x + i + 4);
    _Float16 h8[8] = {(_Float16)a.x,(_Float16)a.y,(_Float16)a.z,(_Float16)a.w,
                      (_Float16)b.x,(_Float16)b.y,(_Float16)b.z,(_Float16)b.w};
    *reinterpret_cast<uint4*>(xh + i) = *reinterpret_cast<uint4*>(h8);
}

// ---------------- 1) gating: logits, top-2 softmax, counts ----------------
__global__ __launch_bounds__(256)
void gate_kernel(const float* __restrict__ x, const float* __restrict__ Wg,
                 const float* __restrict__ bg,
                 int* __restrict__ gidx, float* __restrict__ gval,
                 float* __restrict__ omega, int* __restrict__ counts) {
    __shared__ float WgS[DIN * NE];            // 4 KB, [d][e]
    const int tid = threadIdx.x;
    for (int i = tid; i < DIN * NE; i += 256) WgS[i] = Wg[i];
    __syncthreads();

    const int tLocal = tid >> 3;               // 32 tokens / block
    const int sub    = tid & 7;                // 8 lanes / token
    const int t      = blockIdx.x * 32 + tLocal;

    float acc[NE];
#pragma unroll
    for (int e = 0; e < NE; ++e) acc[e] = 0.f;

    const float* xr = x + (size_t)t * DIN + sub * 16;
#pragma unroll
    for (int j = 0; j < 16; j += 4) {
        const float4 xv = *reinterpret_cast<const float4*>(xr + j);
        const int d = sub * 16 + j;
#pragma unroll
        for (int e = 0; e < NE; ++e) {
            acc[e] += xv.x * WgS[(d + 0) * NE + e]
                    + xv.y * WgS[(d + 1) * NE + e]
                    + xv.z * WgS[(d + 2) * NE + e]
                    + xv.w * WgS[(d + 3) * NE + e];
        }
    }
#pragma unroll
    for (int e = 0; e < NE; ++e) {
        acc[e] += __shfl_xor(acc[e], 1, 32);
        acc[e] += __shfl_xor(acc[e], 2, 32);
        acc[e] += __shfl_xor(acc[e], 4, 32);
    }
    if (sub == 0) {
        float v0 = -INFINITY, v1 = -INFINITY; int i0 = 0, i1 = 0;
#pragma unroll
        for (int e = 0; e < NE; ++e) {
            const float v = acc[e] + bg[e];
            if (v > v0)      { v1 = v0; i1 = i0; v0 = v; i0 = e; }
            else if (v > v1) { v1 = v;  i1 = e; }
        }
        const float z  = __expf(v1 - v0);      // softmax over {v0,v1}
        const float g0 = 1.f / (1.f + z);
        const float g1 = z  / (1.f + z);
        gidx[2 * t + 0] = i0;  gidx[2 * t + 1] = i1;
        gval[2 * t + 0] = g0;  gval[2 * t + 1] = g1;
        omega[t] = 1.f - g0;                   // g0 == max(gating)
        atomicAdd(&counts[i0], 1);
        atomicAdd(&counts[i1], 1);
    }
}

// ---------------- 2) exclusive scan over 8 experts ----------------
__global__ void scan_kernel(const int* __restrict__ counts,
                            int* __restrict__ bases, int* __restrict__ cursor) {
    if (threadIdx.x == 0) {
        int run = 0;
        for (int e = 0; e < NE; ++e) { bases[e] = run; cursor[e] = run; run += counts[e]; }
    }
}

// ---------------- 3) scatter tokens into expert-major order ----------------
__global__ __launch_bounds__(256)
void scatter_kernel(const int* __restrict__ gidx, const float* __restrict__ gval,
                    int* __restrict__ cursor,
                    int* __restrict__ permTok, float* __restrict__ permGate) {
    const int t = blockIdx.x * 256 + threadIdx.x;
#pragma unroll
    for (int s = 0; s < 2; ++s) {
        const int e = gidx[2 * t + s];
        const int p = atomicAdd(&cursor[e], 1);
        permTok[p]  = t;
        permGate[p] = gval[2 * t + s];
    }
}

// ---------------- LDS layout (halves) for FFN kernels ----------------
#define LDS_W1T  0
#define LDS_W2T  (HID * SW_D)
#define LDS_XT0  (LDS_W2T + DIN * SW_H)
#define LDS_XT1  (LDS_XT0 + 16 * SW_D)
#define LDS_HT   (LDS_XT1 + 16 * SW_D)
#define LDS_B1   (LDS_HT + 16 * SW_H)                 // halves; fp32 region follows
#define LDS_BYTES (LDS_B1 * 2 + (HID + DIN) * 4)      // = 155,904 B (< 320 KB/WGP)

// async-stage one pre-transposed f16 weight matrix [N][K] into padded LDS
__device__ __forceinline__ void stage_weight(_Float16* dstT, const _Float16* srcT,
                                             int N, int K, int strideLds, int tid,
                                             int log2cpr /* log2(K/8) */) {
    const int chunks = N * (K >> 3);
    for (int c = tid; c < chunks; c += 256) {
        const int n = c >> log2cpr;
        const int seg = c & ((K >> 3) - 1);
        async_ld_b128(dstT + n * strideLds + seg * 8, srcT,
                      (uint32_t)((n * K + seg * 8) * 2));
    }
}

// ---------------- 4) universal expert: out = omega * uo ----------------
__global__ __launch_bounds__(256)
void universal_kernel(const _Float16* __restrict__ xh,
                      const _Float16* __restrict__ Wu1T, const float* __restrict__ bu1,
                      const _Float16* __restrict__ Wu2T, const float* __restrict__ bu2,
                      const float* __restrict__ omega, float* __restrict__ out) {
    extern __shared__ char smem[];
    _Float16* W1T = reinterpret_cast<_Float16*>(smem);
    _Float16* W2T = W1T + LDS_W2T;
    _Float16* XT0 = W1T + LDS_XT0;
    _Float16* XT1 = W1T + LDS_XT1;
    _Float16* HT  = W1T + LDS_HT;
    float*    b1s = reinterpret_cast<float*>(W1T + LDS_B1);
    float*    b2s = b1s + HID;

    const int tid = threadIdx.x;
    stage_weight(W1T, Wu1T, HID, DIN, SW_D, tid, 4);  // [H][D]
    stage_weight(W2T, Wu2T, DIN, HID, SW_H, tid, 5);  // [D][H]
    if (tid < HID) b1s[tid] = bu1[tid];
    if (tid < DIN) b2s[tid] = bu2[tid];

    const int lane = tid & 31, w = tid >> 5;
    const int row = tid >> 4, seg = tid & 15;
    const int nTiles = NTOK / 16;

    // prologue: async-stage this block's first tile (blockIdx.x) into XT0
    async_ld_b128(XT0 + row * SW_D + seg * 8, xh,
                  (uint32_t)(((blockIdx.x * 16 + row) * DIN + seg * 8) * 2));

    int buf = 0;
    for (int tile = blockIdx.x; tile < nTiles; tile += gridDim.x) {
        wait_async0();
        __syncthreads();                       // current XT buffer + weights ready
        _Float16* XTc = buf ? XT1 : XT0;
        const int nxt = tile + gridDim.x;
        if (nxt < nTiles) {                    // overlap next gather with WMMAs
            _Float16* XTn = buf ? XT0 : XT1;
            async_ld_b128(XTn + row * SW_D + seg * 8, xh,
                          (uint32_t)(((nxt * 16 + row) * DIN + seg * 8) * 2));
        }
        const int tb = tile * 16;
        // GEMM1: [16 x 128] x [128 x 256] ; wave w owns columns [w*32, w*32+32)
        v8f c0 = {0.f,0.f,0.f,0.f,0.f,0.f,0.f,0.f}, c1 = c0;
        const int nb0 = w * 32, nb1 = nb0 + 16;
        const _Float16* arow = XTc + (lane & 15) * SW_D;
#pragma unroll
        for (int kb = 0; kb < DIN; kb += 32) {
            const v16h a = loadA(arow, kb, lane);
            c0 = wmma16(a, loadB(W1T, SW_D, nb0, kb, lane), c0);
            c1 = wmma16(a, loadB(W1T, SW_D, nb1, kb, lane), c1);
        }
        {   // relu(c + b1) -> HT
            const int mb = (lane >> 4) << 3;
            const int n0 = nb0 + (lane & 15), n1 = nb1 + (lane & 15);
            const float bb0 = b1s[n0], bb1 = b1s[n1];
#pragma unroll
            for (int r = 0; r < 8; ++r) {
                float v = c0[r] + bb0;  HT[(mb + r) * SW_H + n0] = (_Float16)(v > 0.f ? v : 0.f);
                v       = c1[r] + bb1;  HT[(mb + r) * SW_H + n1] = (_Float16)(v > 0.f ? v : 0.f);
            }
        }
        __syncthreads();
        // GEMM2: [16 x 256] x [256 x 128] ; wave w owns columns [w*16, w*16+16)
        v8f c2 = {0.f,0.f,0.f,0.f,0.f,0.f,0.f,0.f};
        const int nb = w * 16;
        const _Float16* hrow = HT + (lane & 15) * SW_H;
#pragma unroll
        for (int kb = 0; kb < HID; kb += 32)
            c2 = wmma16(loadA(hrow, kb, lane), loadB(W2T, SW_H, nb, kb, lane), c2);
        {
            const int n = nb + (lane & 15);
            const int mb = (lane >> 4) << 3;
            const float bias = b2s[n];
#pragma unroll
            for (int r = 0; r < 8; ++r) {
                const int m = mb + r;
                out[(size_t)(tb + m) * DIN + n] = omega[tb + m] * (c2[r] + bias);
            }
        }
        __syncthreads();                       // HT reusable next iteration
        buf ^= 1;
    }
}

// ---------------- 5) top-2 experts: out += gate * expert(x) ----------------
__global__ __launch_bounds__(256)
void expert_kernel(const _Float16* __restrict__ xh,
                   const _Float16* __restrict__ W1h, const float* __restrict__ b1,
                   const _Float16* __restrict__ W2h, const float* __restrict__ b2,
                   const int* __restrict__ bases, const int* __restrict__ counts,
                   const int* __restrict__ permTok, const float* __restrict__ permGate,
                   float* __restrict__ out) {
    const int e   = blockIdx.y;
    const int cnt = counts[e];
    const int nTiles = (cnt + 15) >> 4;
    if ((int)blockIdx.x >= nTiles) return;

    extern __shared__ char smem[];
    _Float16* W1T = reinterpret_cast<_Float16*>(smem);
    _Float16* W2T = W1T + LDS_W2T;
    _Float16* XT0 = W1T + LDS_XT0;
    _Float16* XT1 = W1T + LDS_XT1;
    _Float16* HT  = W1T + LDS_HT;
    float*    b1s = reinterpret_cast<float*>(W1T + LDS_B1);
    float*    b2s = b1s + HID;

    const int tid = threadIdx.x;
    stage_weight(W1T, W1h + (size_t)e * HID * DIN, HID, DIN, SW_D, tid, 4);
    stage_weight(W2T, W2h + (size_t)e * DIN * HID, DIN, HID, SW_H, tid, 5);
    if (tid < HID) b1s[tid] = b1[e * HID + tid];
    if (tid < DIN) b2s[tid] = b2[e * DIN + tid];

    const int lane = tid & 31, w = tid >> 5;
    const int row = tid >> 4, seg = tid & 15;
    const int bs = bases[e];

    // gather-stage one 16-token tile (invalid rows -> token 0, results masked later)
    auto stageX = [&](int tile, _Float16* XTbuf) {
        const int ie  = tile * 16 + row;
        const int tok = (ie < cnt) ? permTok[bs + ie] : 0;
        async_ld_b128(XTbuf + row * SW_D + seg * 8, xh,
                      (uint32_t)((tok * DIN + seg * 8) * 2));
    };

    stageX(blockIdx.x, XT0);                   // prologue
    int buf = 0;
    for (int tile = blockIdx.x; tile < nTiles; tile += gridDim.x) {
        wait_async0();
        __syncthreads();
        _Float16* XTc = buf ? XT1 : XT0;
        const int nxt = tile + gridDim.x;
        if (nxt < nTiles) stageX(nxt, buf ? XT0 : XT1);

        v8f c0 = {0.f,0.f,0.f,0.f,0.f,0.f,0.f,0.f}, c1 = c0;
        const int nb0 = w * 32, nb1 = nb0 + 16;
        const _Float16* arow = XTc + (lane & 15) * SW_D;
#pragma unroll
        for (int kb = 0; kb < DIN; kb += 32) {
            const v16h a = loadA(arow, kb, lane);
            c0 = wmma16(a, loadB(W1T, SW_D, nb0, kb, lane), c0);
            c1 = wmma16(a, loadB(W1T, SW_D, nb1, kb, lane), c1);
        }
        {
            const int mb = (lane >> 4) << 3;
            const int n0 = nb0 + (lane & 15), n1 = nb1 + (lane & 15);
            const float bb0 = b1s[n0], bb1 = b1s[n1];
#pragma unroll
            for (int r = 0; r < 8; ++r) {
                float v = c0[r] + bb0;  HT[(mb + r) * SW_H + n0] = (_Float16)(v > 0.f ? v : 0.f);
                v       = c1[r] + bb1;  HT[(mb + r) * SW_H + n1] = (_Float16)(v > 0.f ? v : 0.f);
            }
        }
        __syncthreads();
        v8f c2 = {0.f,0.f,0.f,0.f,0.f,0.f,0.f,0.f};
        const int nb = w * 16;
        const _Float16* hrow = HT + (lane & 15) * SW_H;
#pragma unroll
        for (int kb = 0; kb < HID; kb += 32)
            c2 = wmma16(loadA(hrow, kb, lane), loadB(W2T, SW_H, nb, kb, lane), c2);
        {
            const int n = nb + (lane & 15);
            const int mb = (lane >> 4) << 3;
            const float bias = b2s[n];
#pragma unroll
            for (int r = 0; r < 8; ++r) {
                const int ie = tile * 16 + mb + r;
                if (ie < cnt) {
                    const int   tok = permTok[bs + ie];
                    const float g   = permGate[bs + ie];
                    atomicAdd(out + (size_t)tok * DIN + n, (c2[r] + bias) * g);
                }
            }
        }
        __syncthreads();
        buf ^= 1;
    }
}

// ---------------- host-side launch ----------------
extern "C" void kernel_launch(void* const* d_in, const int* in_sizes, int n_in,
                              void* d_out, int out_size, void* d_ws, size_t ws_size,
                              hipStream_t stream) {
    const float* x   = (const float*)d_in[0];
    const float* W1  = (const float*)d_in[1];
    const float* b1  = (const float*)d_in[2];
    const float* W2  = (const float*)d_in[3];
    const float* b2  = (const float*)d_in[4];
    const float* Wu1 = (const float*)d_in[5];
    const float* bu1 = (const float*)d_in[6];
    const float* Wu2 = (const float*)d_in[7];
    const float* bu2 = (const float*)d_in[8];
    const float* Wg  = (const float*)d_in[9];
    const float* bg  = (const float*)d_in[10];
    float* out = (float*)d_out;

    // workspace carve (all regions rewritten every call; no cross-call state)
    char* ws = (char*)d_ws;
    int*      counts   = (int*)ws;               ws += 256;
    int*      bases    = (int*)ws;               ws += 256;
    int*      cursor   = (int*)ws;               ws += 256;
    int*      gidx     = (int*)ws;               ws += (size_t)2 * NTOK * 4;
    float*    gval     = (float*)ws;             ws += (size_t)2 * NTOK * 4;
    float*    omega    = (float*)ws;             ws += (size_t)NTOK * 4;
    int*      permTok  = (int*)ws;               ws += (size_t)2 * NTOK * 4;
    float*    permGate = (float*)ws;             ws += (size_t)2 * NTOK * 4;
    _Float16* W1h      = (_Float16*)ws;          ws += (size_t)NE * DIN * HID * 2;
    _Float16* W2h      = (_Float16*)ws;          ws += (size_t)NE * HID * DIN * 2;
    _Float16* Wu1h     = (_Float16*)ws;          ws += (size_t)DIN * HID * 2;
    _Float16* Wu2h     = (_Float16*)ws;          ws += (size_t)HID * DIN * 2;
    _Float16* xh       = (_Float16*)ws;

    hipMemsetAsync(counts, 0, NE * sizeof(int), stream);
    // prep: f16 conversions (x coalesced; weights transposed to [N][K])
    xcvt_kernel<<<NTOK * DIN / 8 / 256, 256, 0, stream>>>(x, xh);
    transcvt_kernel<<<dim3(DIN * HID / 256, NE), 256, 0, stream>>>(W1, W1h, DIN, HID);
    transcvt_kernel<<<dim3(HID * DIN / 256, NE), 256, 0, stream>>>(W2, W2h, HID, DIN);
    transcvt_kernel<<<dim3(DIN * HID / 256, 1),  256, 0, stream>>>(Wu1, Wu1h, DIN, HID);
    transcvt_kernel<<<dim3(HID * DIN / 256, 1),  256, 0, stream>>>(Wu2, Wu2h, HID, DIN);
    // routing
    gate_kernel<<<NTOK / 32, 256, 0, stream>>>(x, Wg, bg, gidx, gval, omega, counts);
    scan_kernel<<<1, 32, 0, stream>>>(counts, bases, cursor);
    scatter_kernel<<<NTOK / 256, 256, 0, stream>>>(gidx, gval, cursor, permTok, permGate);
    // FFN
    universal_kernel<<<512, 256, LDS_BYTES, stream>>>(xh, Wu1h, bu1, Wu2h, bu2, omega, out);
    expert_kernel<<<dim3(64, NE), 256, LDS_BYTES, stream>>>(
        xh, W1h, b1, W2h, b2, bases, counts, permTok, permGate, out);
}
